// QuantViTIntermediate_2645699854488
// MI455X (gfx1250) — 8-bit matrix operand order repaired and checked by reading, compiled, not run on hardware
//
#include <hip/hip_runtime.h>
#include <math.h>


typedef int   v8i __attribute__((ext_vector_type(8)));
typedef float v4f __attribute__((ext_vector_type(4)));

#define M_DIM 12608
#define K_DIM 1024
#define N_DIM 4096

__device__ __forceinline__ void async_copy_b128(unsigned lds_addr, const void* gaddr) {
  asm volatile("global_load_async_to_lds_b128 %0, %1, off"
               :: "v"(lds_addr), "v"(gaddr)
               : "memory");
}

__device__ __forceinline__ void wait_asynccnt0() {
#if __has_builtin(__builtin_amdgcn_s_wait_asynccnt)
  __builtin_amdgcn_s_wait_asynccnt(0);
#else
  asm volatile("s_wait_asynccnt 0" ::: "memory");
#endif
}

__device__ __forceinline__ unsigned lds_off(const void* p) {
  return (unsigned)(uintptr_t)p;
}

__global__ __launch_bounds__(256)
void quant_act_kernel(const float* __restrict__ x,
                      const float* __restrict__ a_scale,
                      const float* __restrict__ a_zero,
                      int* __restrict__ xq4) {
  const int t = blockIdx.x * 256 + threadIdx.x;
  const float inv = 1.0f / a_scale[0];
  const float zp  = rintf(a_zero[0]);
  const v4f v = __builtin_nontemporal_load((const v4f*)x + t);
  int q0 = (int)fminf(fmaxf(fminf(fmaxf(rintf(v.x * inv) + zp, 0.f), 255.f) - zp, -128.f), 127.f);
  int q1 = (int)fminf(fmaxf(fminf(fmaxf(rintf(v.y * inv) + zp, 0.f), 255.f) - zp, -128.f), 127.f);
  int q2 = (int)fminf(fmaxf(fminf(fmaxf(rintf(v.z * inv) + zp, 0.f), 255.f) - zp, -128.f), 127.f);
  int q3 = (int)fminf(fmaxf(fminf(fmaxf(rintf(v.w * inv) + zp, 0.f), 255.f) - zp, -128.f), 127.f);
  xq4[t] = (q0 & 0xFF) | ((q1 & 0xFF) << 8) | ((q2 & 0xFF) << 16) | ((q3 & 0xFF) << 24);
}

__global__ __launch_bounds__(256)
void quant_wt_kernel(const float* __restrict__ w,
                     const float* __restrict__ w_scale,
                     const float* __restrict__ w_zero,
                     int* __restrict__ wq4) {
  const int t   = blockIdx.x * 256 + threadIdx.x;
  const int row = t >> 8;
  const float inv = 1.0f / w_scale[row];
  const float zp  = rintf(w_zero[row]);
  const v4f v = __builtin_nontemporal_load((const v4f*)w + t);
  int q0 = (int)fminf(fmaxf(fminf(fmaxf(rintf(v.x * inv) + zp, 0.f), 255.f) - zp, -128.f), 127.f);
  int q1 = (int)fminf(fmaxf(fminf(fmaxf(rintf(v.y * inv) + zp, 0.f), 255.f) - zp, -128.f), 127.f);
  int q2 = (int)fminf(fmaxf(fminf(fmaxf(rintf(v.z * inv) + zp, 0.f), 255.f) - zp, -128.f), 127.f);
  int q3 = (int)fminf(fmaxf(fminf(fmaxf(rintf(v.w * inv) + zp, 0.f), 255.f) - zp, -128.f), 127.f);
  wq4[t] = (q0 & 0xFF) | ((q1 & 0xFF) << 8) | ((q2 & 0xFF) << 16) | ((q3 & 0xFF) << 24);
}

__global__ __launch_bounds__(256)
void gemm_i8_wmma_gelu(const signed char* __restrict__ Aq,
                       const signed char* __restrict__ Bq,
                       const float* __restrict__ bias,
                       const float* __restrict__ w_scale,
                       const float* __restrict__ a_scale_p,
                       float* __restrict__ out) {
  __shared__ signed char lA[2][64 * 64];
  __shared__ signed char lB[2][128 * 64];

  const int tid    = threadIdx.x;
  const int lane   = tid & 31;
  const int wave   = tid >> 5;
  const int waveM  = wave & 3;
  const int waveN  = wave >> 2;
  const int laneLo = lane & 15;
  const int laneHi = lane >> 4;

  const int m0 = blockIdx.y * 64;
  const int n0 = blockIdx.x * 128;

  const int ar = tid >> 2;
  const int ac = (tid & 3) << 4;
  const signed char* gA  = Aq + (size_t)(m0 + ar) * K_DIM + ac;
  const signed char* gB0 = Bq + (size_t)(n0 + ar) * K_DIM + ac;
  const signed char* gB1 = Bq + (size_t)(n0 + ar + 64) * K_DIM + ac;

  const unsigned dA[2]  = { lds_off(&lA[0][ar * 64 + ac]),
                            lds_off(&lA[1][ar * 64 + ac]) };
  const unsigned dB0[2] = { lds_off(&lB[0][ar * 64 + ac]),
                            lds_off(&lB[1][ar * 64 + ac]) };
  const unsigned dB1[2] = { lds_off(&lB[0][(ar + 64) * 64 + ac]),
                            lds_off(&lB[1][(ar + 64) * 64 + ac]) };

  const v8i vzero = {0, 0, 0, 0, 0, 0, 0, 0};
  v8i acc[4];
  acc[0] = vzero; acc[1] = vzero; acc[2] = vzero; acc[3] = vzero;

  async_copy_b128(dA[0],  gA);
  async_copy_b128(dB0[0], gB0);
  async_copy_b128(dB1[0], gB1);
  wait_asynccnt0();
  __syncthreads();

#pragma unroll
  for (int ks = 0; ks < 16; ++ks) {
    const int  buf  = ks & 1;
    const bool more = (ks + 1) < 16;
    if (more) {
      const int koff = (ks + 1) * 64;
      async_copy_b128(dA[buf ^ 1],  gA + koff);
      async_copy_b128(dB0[buf ^ 1], gB0 + koff);
      async_copy_b128(dB1[buf ^ 1], gB1 + koff);
    }

    union { v8i v; unsigned long long q[4]; } au;
    const signed char* arow = &lA[buf][(waveM * 16 + laneLo) * 64 + laneHi * 16];
    au.q[0] = *(const unsigned long long*)(arow + 0);
    au.q[1] = *(const unsigned long long*)(arow + 8);
    au.q[2] = *(const unsigned long long*)(arow + 32);
    au.q[3] = *(const unsigned long long*)(arow + 40);

    union { v8i v; int4 x[2]; } bu[4];
#pragma unroll
    for (int j = 0; j < 4; ++j) {
      const signed char* brow =
          &lB[buf][(waveN * 64 + j * 16 + laneLo) * 64 + laneHi * 16];
      bu[j].x[0] = *(const int4*)(brow);
      bu[j].x[1] = *(const int4*)(brow + 32);
    }
#pragma unroll
    for (int j = 0; j < 4; ++j) {
      acc[j] = __builtin_amdgcn_wmma_i32_16x16x64_iu8(
           true, au.v,  true, bu[j].v, acc[j],
           false,  false);
    }

    if (more) wait_asynccnt0();
    __syncthreads();
  }

  const float as    = a_scale_p[0];
  const int   mBase = m0 + waveM * 16 + laneHi * 8;
#pragma unroll
  for (int j = 0; j < 4; ++j) {
    const int   n = n0 + waveN * 64 + j * 16 + laneLo;
    const float s = as * w_scale[n];
    const float b = bias[n];
    float* op = out + (size_t)mBase * N_DIM + n;
#pragma unroll
    for (int e = 0; e < 8; ++e) {
      const float y = (float)acc[j][e] * s + b;
      const float g = 0.5f * y * (1.0f + erff(y * 0.70710678118654752f));
      __builtin_nontemporal_store(g, op + (size_t)e * N_DIM);
    }
  }
}

extern "C" void kernel_launch(void* const* d_in, const int* in_sizes, int n_in,
                              void* d_out, int out_size, void* d_ws, size_t ws_size,
                              hipStream_t stream) {
  const float* hs      = (const float*)d_in[0];
  const float* weight  = (const float*)d_in[1];
  const float* bias    = (const float*)d_in[2];
  const float* w_scale = (const float*)d_in[3];
  const float* w_zero  = (const float*)d_in[4];
  const float* a_scale = (const float*)d_in[5];
  const float* a_zero  = (const float*)d_in[6];
  float* out = (float*)d_out;

  signed char* xq = (signed char*)d_ws;
  signed char* wq = xq + (size_t)M_DIM * K_DIM;

  quant_act_kernel<<<(M_DIM * K_DIM / 4) / 256, 256, 0, stream>>>(
      hs, a_scale, a_zero, (int*)xq);
  quant_wt_kernel<<<(N_DIM * K_DIM / 4) / 256, 256, 0, stream>>>(
      weight, w_scale, w_zero, (int*)wq);

  dim3 grid(N_DIM / 128, M_DIM / 64);
  gemm_i8_wmma_gelu<<<grid, 256, 0, stream>>>(xq, wq, bias, w_scale, a_scale, out);
}
